// Discriminator_62448824484405
// MI455X (gfx1250) — matrix operand order repaired and checked by reading, not compiled, not run on hardware
//
#include <hip/hip_runtime.h>
#include <hip/hip_bf16.h>

typedef __bf16 bf16_t;
typedef __attribute__((ext_vector_type(16))) __bf16 v16bf;
typedef __attribute__((ext_vector_type(8)))  __bf16 v8bf;
typedef __attribute__((ext_vector_type(8)))  float  v8f;

#define WMMA_BF16(a, b, c) \
    __builtin_amdgcn_wmma_f32_16x16x32_bf16(false, (a), false, (b), (short)0, (c), false, false)

static constexpr int Bv = 64, Tv = 512, Iv = 256, Hv = 512;
static constexpr int Mrows = Bv * Tv;  // 32768

// ---------------------------------------------------------------------------
// fp32 -> bf16 conversion (grid-stride)
// ---------------------------------------------------------------------------
__global__ void k_cvt_bf16(const float* __restrict__ in, bf16_t* __restrict__ out, int n) {
    int i = blockIdx.x * blockDim.x + threadIdx.x;
    int stride = gridDim.x * blockDim.x;
    for (; i < n; i += stride) out[i] = (bf16_t)in[i];
}

// ---------------------------------------------------------------------------
// Projection GEMM: out[M,512] = A[M,K](bf16) * W[512,K]^T (bf16) + b1 + b2
// One wave -> 16x64 output strip (4 accumulators, A reused 4x per k-step).
// ---------------------------------------------------------------------------
__global__ void __launch_bounds__(256) k_proj(const bf16_t* __restrict__ A,
                                              const bf16_t* __restrict__ W,
                                              const float* __restrict__ b1,
                                              const float* __restrict__ b2,
                                              float* __restrict__ out, int K) {
    const int lane = threadIdx.x & 31;
    const int wave = threadIdx.x >> 5;
    const int tile = blockIdx.x * 8 + wave;
    const int tm = tile >> 3, tn4 = tile & 7;
    const int m0 = tm * 16, n0 = tn4 * 64;
    const int ln = lane & 15, half = lane >> 4;

    v8f c[4];
#pragma unroll
    for (int j = 0; j < 4; j++) c[j] = (v8f){0.f, 0.f, 0.f, 0.f, 0.f, 0.f, 0.f, 0.f};

    const bf16_t* arow = A + (size_t)(m0 + ln) * K;
    for (int kb = 0; kb < K; kb += 32) {
        v16bf a;
        ((v8bf*)&a)[0] = *(const v8bf*)(arow + kb + 16 * half);
        ((v8bf*)&a)[1] = *(const v8bf*)(arow + kb + 16 * half + 8);
#pragma unroll
        for (int j = 0; j < 4; j++) {
            const bf16_t* wp = W + (size_t)(n0 + j * 16 + ln) * K + kb + 16 * half;
            v16bf b;
            ((v8bf*)&b)[0] = *(const v8bf*)wp;
            ((v8bf*)&b)[1] = *(const v8bf*)(wp + 8);
            c[j] = WMMA_BF16(a, b, c[j]);
        }
    }
#pragma unroll
    for (int j = 0; j < 4; j++) {
        const int n = n0 + j * 16 + ln;
        const float bias = b1[n] + b2[n];
#pragma unroll
        for (int r = 0; r < 8; r++) {
            out[(size_t)(m0 + 8 * half + r) * Hv + n] = c[j][r] + bias;
        }
    }
}

// ---------------------------------------------------------------------------
// Recurrent scan: h_t = relu(xw[:,t,:] + h_{t-1} @ W_hh^T)
// Block = one 16-row batch slab (4 blocks), 1024 threads = 32 waves; wave w
// owns a 16-column tile. h double-buffered in LDS -> ONE barrier per step.
// W_hh: first 8 k-slices register-resident (64 VGPRs/wave), rest streamed
// from L2. xw prefetched to registers before the WMMA chain. h published to
// global via global_store_async_from_lds_b128 (ASYNCcnt), off the critical
// path; each wave stores exactly the strip it later overwrites, so a
// per-wave s_wait_asynccnt suffices (no extra barrier).
// ---------------------------------------------------------------------------
__global__ void __launch_bounds__(1024, 1) k_rnn(const float* __restrict__ xw,
                                                 const bf16_t* __restrict__ Whh,
                                                 bf16_t* __restrict__ hout) {
    __shared__ bf16_t hb[2][16][Hv + 8];  // double-buffered h, padded rows
    const int lane = threadIdx.x & 31;
    const int wave = threadIdx.x >> 5;  // 0..31 -> column tile
    const int ln = lane & 15, half = lane >> 4;
    const int m0 = blockIdx.x * 16;  // batch-row base
    const int n0 = wave * 16;
    const int srow = lane >> 1;            // async publish: row in strip
    const int scol = n0 + 8 * (lane & 1);  // async publish: col chunk (8 bf16)
    const unsigned long long houtBase = (unsigned long long)(uintptr_t)hout;

    // Register-resident W_hh slice for K = 0..255 (8 k-steps x 8 VGPRs).
    v16bf wreg[8];
#pragma unroll
    for (int ks = 0; ks < 8; ks++) {
        const bf16_t* wp = Whh + (size_t)(n0 + ln) * Hv + ks * 32 + 16 * half;
        ((v8bf*)&wreg[ks])[0] = *(const v8bf*)wp;
        ((v8bf*)&wreg[ks])[1] = *(const v8bf*)(wp + 8);
    }

    for (int i = threadIdx.x; i < 16 * (Hv + 8); i += 1024)
        ((bf16_t*)hb[0])[i] = (bf16_t)0.0f;
    __syncthreads();

    for (int t = 0; t < Tv; t++) {
        const bf16_t(*rd)[Hv + 8] = hb[t & 1];        // h_{t-1}
        bf16_t(*wr)[Hv + 8] = hb[(t + 1) & 1];        // h_t

        // Publish h_{t-1} (still live in rd) to global via async LDS->mem DMA.
        if (t > 0) {
            unsigned voff = (unsigned)((((unsigned)(m0 + srow) * Tv + (t - 1)) * Hv + scol) *
                                       (unsigned)sizeof(bf16_t));
            unsigned lds = (unsigned)(uintptr_t)&rd[srow][scol];
            asm volatile("global_store_async_from_lds_b128 %0, %1, %2"
                         :: "v"(voff), "v"(lds), "s"(houtBase)
                         : "memory");
        }

        // Prefetch this step's xw tile (independent of h).
        float xv[8];
#pragma unroll
        for (int r = 0; r < 8; r++)
            xv[r] = xw[((size_t)(m0 + 8 * half + r) * Tv + t) * Hv + n0 + ln];

        v8f c = (v8f){0.f, 0.f, 0.f, 0.f, 0.f, 0.f, 0.f, 0.f};
#pragma unroll
        for (int ks = 0; ks < 8; ks++) {  // register-resident W half
            const int kb = ks * 32;
            v16bf a;
            ((v8bf*)&a)[0] = *(const v8bf*)&rd[ln][kb + 16 * half];
            ((v8bf*)&a)[1] = *(const v8bf*)&rd[ln][kb + 16 * half + 8];
            c = WMMA_BF16(a, wreg[ks], c);
        }
#pragma unroll 4
        for (int ks = 8; ks < Hv / 32; ks++) {  // streamed W half (L2-hot)
            const int kb = ks * 32;
            v16bf a;
            ((v8bf*)&a)[0] = *(const v8bf*)&rd[ln][kb + 16 * half];
            ((v8bf*)&a)[1] = *(const v8bf*)&rd[ln][kb + 16 * half + 8];
            const bf16_t* wp = Whh + (size_t)(n0 + ln) * Hv + kb + 16 * half;
            v16bf b;
            ((v8bf*)&b)[0] = *(const v8bf*)wp;
            ((v8bf*)&b)[1] = *(const v8bf*)(wp + 8);
            c = WMMA_BF16(a, b, c);
        }

        // The async store issued at step t-1 read what is now `wr`; make sure
        // this wave's outstanding async stores finished before overwriting.
        asm volatile("s_wait_asynccnt 0" ::: "memory");

#pragma unroll
        for (int r = 0; r < 8; r++) {
            const int m = 8 * half + r;
            float v = c[r] + xv[r];
            v = v > 0.f ? v : 0.f;
            wr[m][n0 + ln] = (bf16_t)v;
        }
        __syncthreads();  // h_t visible; all reads of rd complete
    }

    // Flush the last timestep (h_{T-1} lives in hb[Tv & 1] == hb[0]).
    {
        const bf16_t(*fb)[Hv + 8] = hb[Tv & 1];
        v8bf vdat = *(const v8bf*)&fb[srow][scol];
        *(v8bf*)(hout + ((size_t)(m0 + srow) * Tv + (Tv - 1)) * Hv + scol) = vdat;
    }
}

// ---------------------------------------------------------------------------
// FC + sigmoid: out[b,t] = sigmoid(dot(h1[b,t,:], W_fc) + b_fc)
// ---------------------------------------------------------------------------
__global__ void __launch_bounds__(256) k_fc(const bf16_t* __restrict__ h,
                                            const float* __restrict__ wfc,
                                            const float* __restrict__ bfc,
                                            float* __restrict__ out) {
    const int lane = threadIdx.x & 31;
    const int wave = threadIdx.x >> 5;
    const int idx = blockIdx.x * 8 + wave;  // 0..32767
    const bf16_t* hp = h + (size_t)idx * Hv + lane * 16;
    const float* wp = wfc + lane * 16;
    float s = 0.f;
#pragma unroll
    for (int i = 0; i < 16; i++) s += (float)hp[i] * wp[i];
#pragma unroll
    for (int off = 16; off > 0; off >>= 1) s += __shfl_down(s, off, 32);
    if (lane == 0) out[idx] = 1.f / (1.f + __expf(-(s + bfc[0])));
}

// ---------------------------------------------------------------------------
extern "C" void kernel_launch(void* const* d_in, const int* in_sizes, int n_in,
                              void* d_out, int out_size, void* d_ws, size_t ws_size,
                              hipStream_t stream) {
    (void)in_sizes; (void)n_in; (void)out_size; (void)ws_size;
    const float* x     = (const float*)d_in[0];
    const float* W_ih0 = (const float*)d_in[1];
    const float* W_hh0 = (const float*)d_in[2];
    const float* b_ih0 = (const float*)d_in[3];
    const float* b_hh0 = (const float*)d_in[4];
    const float* W_ih1 = (const float*)d_in[5];
    const float* W_hh1 = (const float*)d_in[6];
    const float* b_ih1 = (const float*)d_in[7];
    const float* b_hh1 = (const float*)d_in[8];
    const float* W_fc  = (const float*)d_in[9];
    const float* b_fc  = (const float*)d_in[10];
    float* out = (float*)d_out;

    size_t off = 0;
    auto carve = [&](size_t bytes) -> char* {
        char* p = (char*)d_ws + off;
        off += (bytes + 255) & ~(size_t)255;
        return p;
    };
    bf16_t* xb    = (bf16_t*)carve((size_t)Mrows * Iv * sizeof(bf16_t));  // 16 MB
    bf16_t* wih0b = (bf16_t*)carve((size_t)Hv * Iv * sizeof(bf16_t));
    bf16_t* whh0b = (bf16_t*)carve((size_t)Hv * Hv * sizeof(bf16_t));
    bf16_t* wih1b = (bf16_t*)carve((size_t)Hv * Hv * sizeof(bf16_t));
    bf16_t* whh1b = (bf16_t*)carve((size_t)Hv * Hv * sizeof(bf16_t));
    float*  xw    = (float*)carve((size_t)Mrows * Hv * sizeof(float));    // 64 MB (reused)
    bf16_t* h0    = (bf16_t*)carve((size_t)Mrows * Hv * sizeof(bf16_t));  // 32 MB
    bf16_t* h1    = (bf16_t*)carve((size_t)Mrows * Hv * sizeof(bf16_t));  // 32 MB

    k_cvt_bf16<<<1024, 256, 0, stream>>>(x, xb, Mrows * Iv);
    k_cvt_bf16<<<256, 256, 0, stream>>>(W_ih0, wih0b, Hv * Iv);
    k_cvt_bf16<<<256, 256, 0, stream>>>(W_hh0, whh0b, Hv * Hv);
    k_cvt_bf16<<<256, 256, 0, stream>>>(W_ih1, wih1b, Hv * Hv);
    k_cvt_bf16<<<256, 256, 0, stream>>>(W_hh1, whh1b, Hv * Hv);

    k_proj<<<2048, 256, 0, stream>>>(xb, wih0b, b_ih0, b_hh0, xw, Iv);
    k_rnn<<<4, 1024, 0, stream>>>(xw, whh0b, h0);

    k_proj<<<2048, 256, 0, stream>>>(h0, wih1b, b_ih1, b_hh1, xw, Hv);
    k_rnn<<<4, 1024, 0, stream>>>(xw, whh1b, h1);

    k_fc<<<4096, 256, 0, stream>>>(h1, W_fc, b_fc, out);
}
